// TransformerCrossEncoderLayer_72911364817066
// MI455X (gfx1250) — matrix operand order repaired and checked by reading, not compiled, not run on hardware
//
#include <hip/hip_runtime.h>
#include <cmath>

typedef __attribute__((ext_vector_type(16))) _Float16 v16h;
typedef __attribute__((ext_vector_type(8)))  _Float16 v8h;
typedef __attribute__((ext_vector_type(8)))  float    v8f;

static constexpr int Bb  = 4;
static constexpr int SEQ = 1024;
static constexpr int CH  = 512;
static constexpr int HID = 2048;
static constexpr int NH  = 8;
static constexpr int HD  = 64;
static constexpr int NB  = 81;      // 9*9 relative-position buckets
static constexpr int MR  = Bb * SEQ; // 4096 rows

union V16H { v16h v; v8h h[2]; };

// ---------------- WMMA fragment helpers (ISA 7.12.2 wave32 layouts) --------
// All offsets are 32-bit so loads use GVS addressing (SGPR base + VGPR i32 offset).
// A-matrix 16x32 f16: lanes 0-15 row M=lane, K pairs {0..7,16..23}; lanes 16-31 K {8..15,24..31}
__device__ __forceinline__ v16h load_fragA(const _Float16* base, int row0, int ld, int k0) {
  int lane = threadIdx.x & 31;
  int off = (row0 + (lane & 15)) * ld + k0 + ((lane >> 4) << 3);
  V16H f;
  f.h[0] = *(const v8h*)(base + off);
  f.h[1] = *(const v8h*)(base + off + 16);
  return f.v;
}
// B-matrix 32x16 f16 from transposed storage BT[n][k]: lanes 0-15 K {0..7,16..23}; lanes 16-31 K {8..15,24..31}
__device__ __forceinline__ v16h load_fragB(const _Float16* base, int col0, int ld, int k0) {
  int lane = threadIdx.x & 31;
  int off = (col0 + (lane & 15)) * ld + k0 + ((lane >> 4) << 3);
  V16H f;
  f.h[0] = *(const v8h*)(base + off);
  f.h[1] = *(const v8h*)(base + off + 16);
  return f.v;
}
__device__ __forceinline__ v8f wmma_f16(v16h a, v16h b, v8f c) {
  return __builtin_amdgcn_wmma_f32_16x16x32_f16(false, a, false, b, (short)0, c, false, false);
}

// ---------------- bucket ids (piecewise log-bucketing) ---------------------
__device__ __forceinline__ int piecewise_idx(float d) {
  const float ALPHA = 1.9f, BETA = 3.8f, GAMMA = 15.2f;
  float ad = fabsf(d);
  float t  = ALPHA + logf(fmaxf(ad, ALPHA) / ALPHA) / logf(GAMMA / ALPHA) * (BETA - ALPHA);
  float lp = copysignf(rintf(fminf(t, BETA)), d);
  float idx = (ad <= ALPHA) ? rintf(d) : lp;
  idx = fminf(fmaxf(idx, -4.f), 4.f);
  return (int)idx;
}

__global__ __launch_bounds__(256) void bucket_kernel(int* __restrict__ bucket) {
  int idx = blockIdx.x * 256 + threadIdx.x;
  if (idx >= SEQ * SEQ) return;
  int i = idx >> 10, j = idx & 1023;
  float dy = (float)((i >> 5) - (j >> 5));
  float dx = (float)((i & 31) - (j & 31));
  bucket[idx] = (piecewise_idx(dy) + 4) * 9 + (piecewise_idx(dx) + 4);
}

// ---------------- fp32 -> fp16 convert (plain / transposing) ---------------
__global__ __launch_bounds__(256) void cvt16_kernel(const float* __restrict__ in,
                                                    _Float16* __restrict__ out, int n) {
  int i = blockIdx.x * 256 + threadIdx.x;
  if (i < n) out[i] = (_Float16)in[i];
}
__global__ __launch_bounds__(256) void transpose16_kernel(const float* __restrict__ w,
                                                          _Float16* __restrict__ wT,
                                                          int K, int N) {
  int idx = blockIdx.x * 256 + threadIdx.x;
  if (idx >= K * N) return;
  int n = idx / K, k = idx % K;
  wT[idx] = (_Float16)w[k * N + n];
}

// ---------------- LayerNorm row kernel (fp32 in, fp16 out) -----------------
__global__ __launch_bounds__(256) void ln_kernel(const float* __restrict__ x,
                                                 const float* __restrict__ g,
                                                 const float* __restrict__ b,
                                                 _Float16* __restrict__ out) {
  int row = blockIdx.x, tid = threadIdx.x;
  const float* xr = x + row * CH;
  float s = 0.f, ss = 0.f;
  #pragma unroll
  for (int c = tid; c < CH; c += 256) { float v = xr[c]; s += v; ss += v * v; }
  #pragma unroll
  for (int m = 16; m >= 1; m >>= 1) { s += __shfl_xor(s, m, 32); ss += __shfl_xor(ss, m, 32); }
  __shared__ float red[16];
  int wave = tid >> 5, lane = tid & 31;
  if (lane == 0) { red[wave] = s; red[wave + 8] = ss; }
  __syncthreads();
  if (tid == 0) {
    float ts = 0.f, tss = 0.f;
    #pragma unroll
    for (int w = 0; w < 8; ++w) { ts += red[w]; tss += red[w + 8]; }
    red[0] = ts; red[8] = tss;
  }
  __syncthreads();
  float mean = red[0] / CH;
  float var  = red[8] / CH - mean * mean;
  float inv  = rsqrtf(var + 1e-5f);
  #pragma unroll
  for (int c = tid; c < CH; c += 256)
    out[row * CH + c] = (_Float16)((xr[c] - mean) * inv * g[c] + b[c]);
}

// ---------------- generic WMMA GEMM: out = act(A @ BT^T + bias) [+res] -----
// A [M,K] f16 row-major, BT [N,K] f16 (pre-transposed). Block tile 128x64, 8 waves.
template <int ACT, int RES, int OUT16>
__global__ __launch_bounds__(256) void gemm_kernel(const _Float16* __restrict__ A,
                                                   const _Float16* __restrict__ BT,
                                                   const float* __restrict__ bias,
                                                   const float* __restrict__ resid,
                                                   float* __restrict__ out32,
                                                   _Float16* __restrict__ out16,
                                                   int M, int Nn, int K) {
  int wave = threadIdx.x >> 5, lane = threadIdx.x & 31;
  int m0 = blockIdx.x * 128 + (wave >> 1) * 32;
  int n0 = blockIdx.y * 64  + (wave & 1)  * 32;
  v8f acc[2][2];
  #pragma unroll
  for (int j = 0; j < 2; ++j) {
    float bv = bias[n0 + j * 16 + (lane & 15)];
    #pragma unroll
    for (int i = 0; i < 2; ++i)
      #pragma unroll
      for (int v = 0; v < 8; ++v) acc[i][j][v] = bv;
  }
  #pragma unroll 4
  for (int k0 = 0; k0 < K; k0 += 32) {
    v16h a0 = load_fragA(A, m0,      K, k0);
    v16h a1 = load_fragA(A, m0 + 16, K, k0);
    v16h b0 = load_fragB(BT, n0,      K, k0);
    v16h b1 = load_fragB(BT, n0 + 16, K, k0);
    acc[0][0] = wmma_f16(a0, b0, acc[0][0]);
    acc[0][1] = wmma_f16(a0, b1, acc[0][1]);
    acc[1][0] = wmma_f16(a1, b0, acc[1][0]);
    acc[1][1] = wmma_f16(a1, b1, acc[1][1]);
  }
  int hi8 = (lane >> 4) << 3;
  #pragma unroll
  for (int i = 0; i < 2; ++i)
    #pragma unroll
    for (int j = 0; j < 2; ++j)
      #pragma unroll
      for (int v = 0; v < 8; ++v) {
        int m = m0 + i * 16 + v + hi8;
        int n = n0 + j * 16 + (lane & 15);
        float x = acc[i][j][v];
        if (ACT == 1) x = 0.5f * x * (1.f + erff(x * 0.70710678118f)); // exact gelu
        if (RES) x += resid[m * Nn + n];
        if (OUT16) out16[m * Nn + n] = (_Float16)x;
        else       out32[m * Nn + n] = x;
      }
}

// ---------------- bias tables: out[b,h,i,c] = scl * <row_i(q/k), tab_c> ----
__global__ __launch_bounds__(96) void bias_tab_kernel(const _Float16* __restrict__ qkv16,
                                                      int colOff,
                                                      const _Float16* __restrict__ tab,
                                                      float scl, float* __restrict__ out) {
  int c = threadIdx.x;
  if (c >= NB) return;
  int bi = blockIdx.x;          // b*SEQ + i
  int h  = blockIdx.y;
  int b  = bi >> 10, i = bi & 1023;
  const _Float16* qp = qkv16 + bi * (3 * CH) + colOff + h * HD;
  const _Float16* tp = tab + c * HD;
  float acc = 0.f;
  #pragma unroll
  for (int d8 = 0; d8 < 8; ++d8) {
    v8h qv = *(const v8h*)(qp + d8 * 8);
    v8h tv = *(const v8h*)(tp + d8 * 8);
    #pragma unroll
    for (int u = 0; u < 8; ++u) acc += (float)qv[u] * (float)tv[u];
  }
  out[((b * NH + h) * SEQ + i) * NB + c] = acc * scl;
}

// ---------------- fused attention (flash-style, bucket-histogram tv term) --
__global__ __launch_bounds__(256) void attn_kernel(const _Float16* __restrict__ qkvQ,
                                                   const _Float16* __restrict__ qkvKV,
                                                   const float* __restrict__ lk,
                                                   const float* __restrict__ lqs,
                                                   const int* __restrict__ bucket,
                                                   const _Float16* __restrict__ tv16,
                                                   _Float16* __restrict__ outA) {
  __shared__ __align__(32) _Float16 Kt[32][64];       // K tile, natural [j][d]
  __shared__ __align__(32) _Float16 VtT[64][32];      // V tile transposed [d][j]
  __shared__ float sbuck[128][NB];                    // per-q-row bucket sums of p
  __shared__ __align__(32) _Float16 Pscr[8][16][32];  // per-wave P re-layout scratch

  int tid = threadIdx.x, wave = tid >> 5, lane = tid & 31;
  int b = blockIdx.z, h = blockIdx.y, q0 = blockIdx.x * 128;
  int hi8 = (lane >> 4) << 3;

  for (int t = lane; t < 16 * NB; t += 32) sbuck[wave * 16 + t / NB][t % NB] = 0.f;

  const _Float16* qbase = qkvQ + (b * SEQ + q0 + wave * 16) * (3 * CH) + h * HD;
  v16h aq0 = load_fragA(qbase, 0, 3 * CH, 0);
  v16h aq1 = load_fragA(qbase, 0, 3 * CH, 32);
  v8f oa[4];
  #pragma unroll
  for (int t = 0; t < 4; ++t)
    #pragma unroll
    for (int v = 0; v < 8; ++v) oa[t][v] = 0.f;
  float lacc[8];
  #pragma unroll
  for (int v = 0; v < 8; ++v) lacc[v] = 0.f;

  const float* lkrow = lk  + (b * NH + h) * SEQ * NB;  // uniform SGPR bases
  const float* lqrow = lqs + (b * NH + h) * SEQ * NB;
  int irow = q0 + wave * 16 + hi8;   // this lane's first q row (v adds 0..7)

  for (int kt = 0; kt < SEQ / 32; ++kt) {
    __syncthreads();
    { // cooperative stage: K natural, V transposed into LDS
      int r = tid >> 3, c8 = (tid & 7) * 8;
      const _Float16* kp = qkvKV + (b * SEQ + kt * 32 + r) * (3 * CH) + CH + h * HD + c8;
      *(v8h*)&Kt[r][c8] = *(const v8h*)kp;
      v8h vv = *(const v8h*)(kp + CH);
      #pragma unroll
      for (int u = 0; u < 8; ++u) VtT[c8 + u][r] = vv[u];
    }
    __syncthreads();

    // phase 1: issue all independent bucket-index loads FIRST so their latency
    // hides under the S-tile WMMAs below
    int cc[2][8], cc2[2][8];
    #pragma unroll
    for (int jt = 0; jt < 2; ++jt) {
      int jn = kt * 32 + jt * 16 + (lane & 15);
      #pragma unroll
      for (int v = 0; v < 8; ++v) {
        int i = irow + v;
        cc[jt][v]  = bucket[i * SEQ + jn];   // bucket[i,j]
        cc2[jt][v] = bucket[jn * SEQ + i];   // bucket[j,i] (vectorizes to b128)
      }
    }

    v8f s01[2];
    #pragma unroll
    for (int v = 0; v < 8; ++v) { s01[0][v] = 0.f; s01[1][v] = 0.f; }
    v16h bk;
    bk = load_fragB(&Kt[0][0], 0, 64, 0);   s01[0] = wmma_f16(aq0, bk, s01[0]);
    bk = load_fragB(&Kt[0][0], 0, 64, 32);  s01[0] = wmma_f16(aq1, bk, s01[0]);
    bk = load_fragB(&Kt[0][0], 16, 64, 0);  s01[1] = wmma_f16(aq0, bk, s01[1]);
    bk = load_fragB(&Kt[0][0], 16, 64, 32); s01[1] = wmma_f16(aq1, bk, s01[1]);

    // phase 2: two independent gather batches (16 lk + 16 lq loads in flight)
    float bkg[2][8], bqg[2][8];
    #pragma unroll
    for (int jt = 0; jt < 2; ++jt) {
      #pragma unroll
      for (int v = 0; v < 8; ++v)
        bkg[jt][v] = lkrow[(irow + v) * NB + cc[jt][v]];
    }
    #pragma unroll
    for (int jt = 0; jt < 2; ++jt) {
      int jbase = (kt * 32 + jt * 16 + (lane & 15)) * NB;
      #pragma unroll
      for (int v = 0; v < 8; ++v)
        bqg[jt][v] = lqrow[jbase + cc2[jt][v]];
    }
    // phase 3: exp, l-accum, bucket histogram, P re-layout
    #pragma unroll
    for (int jt = 0; jt < 2; ++jt) {
      #pragma unroll
      for (int v = 0; v < 8; ++v) {
        int m = v + hi8;
        float p = __expf(s01[jt][v] * 0.125f + bkg[jt][v] + bqg[jt][v]);
        lacc[v] += p;
        atomicAdd(&sbuck[wave * 16 + m][cc[jt][v]], p);    // ds_add_f32
        Pscr[wave][m][jt * 16 + (lane & 15)] = (_Float16)p;
      }
    }
    asm volatile("s_wait_dscnt 0" ::: "memory");
    v16h ap = load_fragA(&Pscr[wave][0][0], 0, 32, 0);
    #pragma unroll
    for (int t = 0; t < 4; ++t) {
      v16h bv = load_fragB(&VtT[0][0], t * 16, 32, 0);
      oa[t] = wmma_f16(ap, bv, oa[t]);
    }
  }

  #pragma unroll
  for (int m = 1; m <= 8; m <<= 1)
    #pragma unroll
    for (int v = 0; v < 8; ++v) lacc[v] += __shfl_xor(lacc[v], m, 32);
  float linv[8];
  #pragma unroll
  for (int v = 0; v < 8; ++v) linv[v] = 1.0f / lacc[v];   // hoisted normalization
  asm volatile("s_wait_dscnt 0" ::: "memory");

  // tv-bucket epilogue, c-outer to reuse loaded values across all 32 outputs
  float tvterm[4][8];
  #pragma unroll
  for (int t = 0; t < 4; ++t)
    #pragma unroll
    for (int v = 0; v < 8; ++v) tvterm[t][v] = 0.f;
  for (int c = 0; c < NB; ++c) {
    float tvv[4];
    #pragma unroll
    for (int t = 0; t < 4; ++t) tvv[t] = (float)tv16[c * HD + t * 16 + (lane & 15)];
    #pragma unroll
    for (int v = 0; v < 8; ++v) {
      float sbv = sbuck[wave * 16 + v + hi8][c];
      #pragma unroll
      for (int t = 0; t < 4; ++t) tvterm[t][v] += sbv * tvv[t];
    }
  }
  #pragma unroll
  for (int t = 0; t < 4; ++t) {
    int col = t * 16 + (lane & 15);
    #pragma unroll
    for (int v = 0; v < 8; ++v) {
      float o = (oa[t][v] + tvterm[t][v]) * linv[v];
      outA[(b * SEQ + irow + v) * CH + h * HD + col] = (_Float16)o;
    }
  }
}

// --------------------------------------------------------------------------
extern "C" void kernel_launch(void* const* d_in, const int* in_sizes, int n_in,
                              void* d_out, int out_size, void* d_ws, size_t ws_size,
                              hipStream_t stream) {
  const float* x0     = (const float*)d_in[0];
  const float* x1     = (const float*)d_in[1];
  const float* ln00_g = (const float*)d_in[2];
  const float* ln00_b = (const float*)d_in[3];
  const float* ln01_g = (const float*)d_in[4];
  const float* ln01_b = (const float*)d_in[5];
  const float* ln10_g = (const float*)d_in[6];
  const float* ln10_b = (const float*)d_in[7];
  const float* ln11_g = (const float*)d_in[8];
  const float* ln11_b = (const float*)d_in[9];
  const float* w_qkv0 = (const float*)d_in[10];
  const float* b_qkv0 = (const float*)d_in[11];
  const float* w_qkv1 = (const float*)d_in[12];
  const float* b_qkv1 = (const float*)d_in[13];
  const float* w_proj = (const float*)d_in[14];
  const float* b_proj = (const float*)d_in[15];
  const float* table_q = (const float*)d_in[16];
  const float* table_k = (const float*)d_in[17];
  const float* table_v = (const float*)d_in[18];
  const float* w_fc1_0 = (const float*)d_in[19];
  const float* b_fc1_0 = (const float*)d_in[20];
  const float* w_fc2_0 = (const float*)d_in[21];
  const float* b_fc2_0 = (const float*)d_in[22];
  const float* w_fc1_1 = (const float*)d_in[23];
  const float* b_fc1_1 = (const float*)d_in[24];
  const float* w_fc2_1 = (const float*)d_in[25];
  const float* b_fc2_1 = (const float*)d_in[26];

  float* out0 = (float*)d_out;
  float* out1 = out0 + (size_t)MR * CH;

  size_t off = 0;
  auto Aw = [&](size_t bytes) -> char* {
    char* p = (char*)d_ws + off;
    off = (off + bytes + 255) & ~(size_t)255;
    return p;
  };
  int*      bucket  = (int*)     Aw((size_t)SEQ * SEQ * 4);
  _Float16* wqkv0T  = (_Float16*)Aw((size_t)CH * 3 * CH * 2);
  _Float16* wqkv1T  = (_Float16*)Aw((size_t)CH * 3 * CH * 2);
  _Float16* wprojT  = (_Float16*)Aw((size_t)CH * CH * 2);
  _Float16* wfc10T  = (_Float16*)Aw((size_t)CH * HID * 2);
  _Float16* wfc20T  = (_Float16*)Aw((size_t)HID * CH * 2);
  _Float16* wfc11T  = (_Float16*)Aw((size_t)CH * HID * 2);
  _Float16* wfc21T  = (_Float16*)Aw((size_t)HID * CH * 2);
  _Float16* tq16    = (_Float16*)Aw((size_t)NB * HD * 2);
  _Float16* tk16    = (_Float16*)Aw((size_t)NB * HD * 2);
  _Float16* tv16    = (_Float16*)Aw((size_t)NB * HD * 2);
  _Float16* xn0     = (_Float16*)Aw((size_t)MR * CH * 2);
  _Float16* xn1     = (_Float16*)Aw((size_t)MR * CH * 2);
  _Float16* qkv0h   = (_Float16*)Aw((size_t)MR * 3 * CH * 2);
  _Float16* qkv1h   = (_Float16*)Aw((size_t)MR * 3 * CH * 2);
  float*    lk_q0   = (float*)   Aw((size_t)Bb * NH * SEQ * NB * 4);
  float*    lk_q1   = (float*)   Aw((size_t)Bb * NH * SEQ * NB * 4);
  float*    lqs_k0  = (float*)   Aw((size_t)Bb * NH * SEQ * NB * 4);
  float*    lqs_k1  = (float*)   Aw((size_t)Bb * NH * SEQ * NB * 4);
  _Float16* at0     = (_Float16*)Aw((size_t)MR * CH * 2);
  _Float16* at1     = (_Float16*)Aw((size_t)MR * CH * 2);
  float*    xmid0   = (float*)   Aw((size_t)MR * CH * 4);
  float*    xmid1   = (float*)   Aw((size_t)MR * CH * 4);
  _Float16* hbuf0   = (_Float16*)Aw((size_t)MR * HID * 2);
  _Float16* hbuf1   = (_Float16*)Aw((size_t)MR * HID * 2);
  (void)ws_size; (void)in_sizes; (void)n_in; (void)out_size;

  // 1) bucket ids
  bucket_kernel<<<(SEQ * SEQ + 255) / 256, 256, 0, stream>>>(bucket);

  // 2) weight transpose+convert to fp16 (L2-resident operand layout for WMMA B-frags)
  auto T = [&](const float* w, _Float16* wT, int K, int N) {
    transpose16_kernel<<<(K * N + 255) / 256, 256, 0, stream>>>(w, wT, K, N);
  };
  T(w_qkv0, wqkv0T, CH, 3 * CH);
  T(w_qkv1, wqkv1T, CH, 3 * CH);
  T(w_proj, wprojT, CH, CH);
  T(w_fc1_0, wfc10T, CH, HID);
  T(w_fc2_0, wfc20T, HID, CH);
  T(w_fc1_1, wfc11T, CH, HID);
  T(w_fc2_1, wfc21T, HID, CH);
  cvt16_kernel<<<(NB * HD + 255) / 256, 256, 0, stream>>>(table_q, tq16, NB * HD);
  cvt16_kernel<<<(NB * HD + 255) / 256, 256, 0, stream>>>(table_k, tk16, NB * HD);
  cvt16_kernel<<<(NB * HD + 255) / 256, 256, 0, stream>>>(table_v, tv16, NB * HD);

  // 3) pre-attention layernorms
  ln_kernel<<<MR, 256, 0, stream>>>(x0, ln00_g, ln00_b, xn0);
  ln_kernel<<<MR, 256, 0, stream>>>(x1, ln01_g, ln01_b, xn1);

  // 4) QKV projections (fp16 WMMA, fp32 accumulate, fp16 out)
  dim3 gq(MR / 128, (3 * CH) / 64);
  gemm_kernel<0, 0, 1><<<gq, 256, 0, stream>>>(xn0, wqkv0T, b_qkv0, nullptr, nullptr, qkv0h, MR, 3 * CH, CH);
  gemm_kernel<0, 0, 1><<<gq, 256, 0, stream>>>(xn1, wqkv1T, b_qkv1, nullptr, nullptr, qkv1h, MR, 3 * CH, CH);

  // 5) relative-position logit tables: lk = q @ tk^T, lqs = scale*(k @ tq^T)
  dim3 gb(Bb * SEQ, NH);
  bias_tab_kernel<<<gb, 96, 0, stream>>>(qkv0h, 0,  tk16, 1.f,    lk_q0);
  bias_tab_kernel<<<gb, 96, 0, stream>>>(qkv1h, 0,  tk16, 1.f,    lk_q1);
  bias_tab_kernel<<<gb, 96, 0, stream>>>(qkv0h, CH, tq16, 0.125f, lqs_k0);
  bias_tab_kernel<<<gb, 96, 0, stream>>>(qkv1h, CH, tq16, 0.125f, lqs_k1);

  // 6) fused cross attention (x0 attends k1/v1; x1 attends k0/v0)
  dim3 ga(SEQ / 128, NH, Bb);
  attn_kernel<<<ga, 256, 0, stream>>>(qkv0h, qkv1h, lk_q0, lqs_k1, bucket, tv16, at0);
  attn_kernel<<<ga, 256, 0, stream>>>(qkv1h, qkv0h, lk_q1, lqs_k0, bucket, tv16, at1);

  // 7) output projection + residual (fp32 out)
  dim3 gp(MR / 128, CH / 64);
  gemm_kernel<0, 1, 0><<<gp, 256, 0, stream>>>(at0, wprojT, b_proj, x0, xmid0, nullptr, MR, CH, CH);
  gemm_kernel<0, 1, 0><<<gp, 256, 0, stream>>>(at1, wprojT, b_proj, x1, xmid1, nullptr, MR, CH, CH);

  // 8) post-attention layernorms (reuse xn buffers)
  ln_kernel<<<MR, 256, 0, stream>>>(xmid0, ln10_g, ln10_b, xn0);
  ln_kernel<<<MR, 256, 0, stream>>>(xmid1, ln11_g, ln11_b, xn1);

  // 9) MLP fc1 + exact GELU (fp16 out)
  dim3 g1(MR / 128, HID / 64);
  gemm_kernel<1, 0, 1><<<g1, 256, 0, stream>>>(xn0, wfc10T, b_fc1_0, nullptr, nullptr, hbuf0, MR, HID, CH);
  gemm_kernel<1, 0, 1><<<g1, 256, 0, stream>>>(xn1, wfc11T, b_fc1_1, nullptr, nullptr, hbuf1, MR, HID, CH);

  // 10) MLP fc2 + residual -> final outputs (fp32, concatenated x0 then x1)
  gemm_kernel<0, 1, 0><<<gp, 256, 0, stream>>>(hbuf0, wfc20T, b_fc2_0, xmid0, out0, nullptr, MR, CH, HID);
  gemm_kernel<0, 1, 0><<<gp, 256, 0, stream>>>(hbuf1, wfc21T, b_fc2_1, xmid1, out1, nullptr, MR, CH, HID);
}
